// GraphSAGE_31001073943304
// MI455X (gfx1250) — matrix operand order repaired and checked by reading, not compiled, not run on hardware
//
#include <hip/hip_runtime.h>
#include <hip/hip_bf16.h>
#include <stdint.h>

// ---------------------------------------------------------------------------
// GraphSAGE (2-layer) for MI455X / gfx1250, wave32, WMMA bf16 16x16x32.
// A/B operands pre-packed to bf16 fragment order so the GEMM kernel is pure
// (b128 loads + v_wmma), no in-loop format conversion.
// ---------------------------------------------------------------------------

#define N_NODES 50000
#define N_EDGES 640000
#define IN_CH   128
#define HID     256
#define OUT_CH  3

typedef __attribute__((ext_vector_type(16))) __bf16          v16bf;
typedef __attribute__((ext_vector_type(16))) unsigned short  v16u;
typedef __attribute__((ext_vector_type(8)))  float           v8f;

__device__ __forceinline__ unsigned short f32_to_bf16_rne(float f) {
    unsigned int u = __float_as_uint(f);
    unsigned int r = u + 0x7FFFu + ((u >> 16) & 1u);   // round-to-nearest-even
    return (unsigned short)(r >> 16);
}

// ---------------------------------------------------------------------------
// Zero a contiguous float region (grid-stride).
// ---------------------------------------------------------------------------
__global__ void zero_f32(float* __restrict__ p, size_t n) {
    size_t i = (size_t)blockIdx.x * blockDim.x + threadIdx.x;
    size_t stride = (size_t)gridDim.x * blockDim.x;
    for (; i < n; i += stride) p[i] = 0.0f;
}

// ---------------------------------------------------------------------------
// Pack a f32 weight matrix W[K=128][N=256] (row-major) into bf16 WMMA
// B-fragment order (B 32x16 per K-tile): n = lane&15, half = lane>>4,
// V_v (v=0..7) packs K = 32*ktile + (v<4?0:16) + 2*(v&3) + 8*half (+0,+1).
// Each lane's 16 bf16 values land as one contiguous 32B block.
// ---------------------------------------------------------------------------
__global__ void pack_weight_bf16(const float* __restrict__ W,
                                 unsigned short* __restrict__ P) {
    int idx = blockIdx.x * blockDim.x + threadIdx.x;   // 4*16*32 = 2048
    if (idx >= 4 * 16 * 32) return;
    int lane  = idx & 31;
    int t     = idx >> 5;
    int ntile = t & 15;
    int kt    = t >> 4;
    int half  = lane >> 4;
    int n     = lane & 15;
    unsigned short* o = P + (size_t)idx * 16;
#pragma unroll
    for (int v = 0; v < 8; ++v) {
        int kb = 32 * kt + ((v < 4) ? 0 : 16) + 2 * (v & 3) + 8 * half;
        o[2 * v]     = f32_to_bf16_rne(W[(size_t)kb       * HID + ntile * 16 + n]);
        o[2 * v + 1] = f32_to_bf16_rne(W[(size_t)(kb + 1) * HID + ntile * 16 + n]);
    }
}

// ---------------------------------------------------------------------------
// Pack a f32 activation matrix feat[N_NODES][128] into bf16 WMMA A-fragment
// order (16-bit A 16x32, ISA 7.12.2): m = lane&15, half = lane>>4,
// V_v packs K = 32*ktile + (v<4?0:16) + 2*(v&3) + 8*half (+0,+1).
// One thread per (rowTile, ktile, lane) writes 16 contiguous bf16 (32B).
// ---------------------------------------------------------------------------
__global__ __launch_bounds__(256) void pack_a_bf16(
    const float* __restrict__ feat, unsigned short* __restrict__ P) {
    int gid = blockIdx.x * blockDim.x + threadIdx.x;   // 3125*4*32 = 400000
    if (gid >= (N_NODES / 16) * 4 * 32) return;
    int lane    = gid & 31;
    int t       = gid >> 5;
    int kt      = t & 3;
    int rowTile = t >> 2;
    int half = lane >> 4;
    int m    = lane & 15;
    const float* row = feat + (size_t)(rowTile * 16 + m) * IN_CH;
    unsigned short* o = P + (size_t)gid * 16;
#pragma unroll
    for (int v = 0; v < 8; ++v) {
        int k0 = 32 * kt + ((v < 4) ? 0 : 16) + 2 * (v & 3) + 8 * half;
        o[2 * v]     = f32_to_bf16_rne(row[k0]);
        o[2 * v + 1] = f32_to_bf16_rne(row[k0 + 1]);
    }
}

// ---------------------------------------------------------------------------
// Edge scatter: msg[dst] += feat[src]  (+ degree on layer 0).
// C/4 lanes per edge, float4 gather, 4x global_atomic_add_f32.
// Accumulators fit in the 192MB L2, so atomics stay on-chip.
// ---------------------------------------------------------------------------
template <int C, bool COUNT_DEG>
__global__ __launch_bounds__(256) void scatter_add(
    const float* __restrict__ feat, const int* __restrict__ eidx,
    float* __restrict__ msg, float* __restrict__ deg, int nEdges) {
    constexpr int LPE = C / 4;
    int gid = blockIdx.x * blockDim.x + threadIdx.x;
    int e = gid / LPE;
    int l = gid % LPE;
    if (e >= nEdges) return;
    int s = eidx[e];
    int d = eidx[nEdges + e];
    const float4 v = *(const float4*)(feat + (size_t)s * C + l * 4);
    float* p = msg + (size_t)d * C + l * 4;
    atomicAdd(p + 0, v.x);
    atomicAdd(p + 1, v.y);
    atomicAdd(p + 2, v.z);
    atomicAdd(p + 3, v.w);
    if (COUNT_DEG && l == 0) atomicAdd(deg + d, 1.0f);
}

// ---------------------------------------------------------------------------
// In-place mean: msg[i] /= max(deg[node], 1)
// ---------------------------------------------------------------------------
template <int SHIFT>   // log2(C)
__global__ void mean_div(float* __restrict__ msg, const float* __restrict__ deg,
                         size_t total) {
    size_t i = (size_t)blockIdx.x * blockDim.x + threadIdx.x;
    if (i >= total) return;
    float d = deg[i >> SHIFT];
    d = d > 1.0f ? d : 1.0f;
    msg[i] = msg[i] / d;
}

// ---------------------------------------------------------------------------
// Layer 0: h = relu(mean0 @ Wl0 + bl0 + x @ Wr0)  via v_wmma_f32_16x16x32_bf16.
// One wave per 16x16 output tile; 50000 tiles = 6250 blocks x 8 waves (exact,
// wave-uniform guard -> EXEC stays all 1s as WMMA requires).
// All operands pre-packed: per K-tile this is 4x 32B fragment loads + 2 WMMAs.
// Both products accumulate into one f32 C chain (D = mean@Wl + x@Wr + bias).
// ---------------------------------------------------------------------------
__global__ __launch_bounds__(256) void gemm0_wmma(
    const unsigned short* __restrict__ AmeanP, const unsigned short* __restrict__ AxP,
    const unsigned short* __restrict__ WlP, const unsigned short* __restrict__ WrP,
    const float* __restrict__ bl, float* __restrict__ h) {
    int wave = threadIdx.x >> 5;
    int lane = threadIdx.x & 31;
    int tile = blockIdx.x * 8 + wave;
    if (tile >= (N_NODES / 16) * 16) return;          // wave-uniform guard
    int rowTile = tile >> 4;
    int ntile   = tile & 15;

    v8f c = {};
#pragma unroll
    for (int kt = 0; kt < 4; ++kt) {
        const size_t aoff = (((size_t)rowTile * 4 + kt) * 32 + lane) * 16;
        const size_t boff = (((size_t)kt * 16 + ntile) * 32 + lane) * 16;
        const v16u am  = *(const v16u*)(AmeanP + aoff);
        const v16u ax  = *(const v16u*)(AxP    + aoff);
        const v16u bfl = *(const v16u*)(WlP + boff);
        const v16u bfr = *(const v16u*)(WrP + boff);
        c = __builtin_amdgcn_wmma_f32_16x16x32_bf16(
                false, __builtin_bit_cast(v16bf, am),
                false, __builtin_bit_cast(v16bf, bfl),
                (short)0, c, false, false);
        c = __builtin_amdgcn_wmma_f32_16x16x32_bf16(
                false, __builtin_bit_cast(v16bf, ax),
                false, __builtin_bit_cast(v16bf, bfr),
                (short)0, c, false, false);
    }

    // C/D layout: VGPR r holds M = r + 8*half, N = lane&15 within tile.
    int half = lane >> 4;
    int col  = ntile * 16 + (lane & 15);
    float bias = bl[col];
#pragma unroll
    for (int r = 0; r < 8; ++r) {
        int nrow = rowTile * 16 + r + 8 * half;
        float v = c[r] + bias;
        h[(size_t)nrow * HID + col] = v > 0.0f ? v : 0.0f;
    }
}

// ---------------------------------------------------------------------------
// Layer 1: out[node][c] = bl1[c] + mean1[node]·Wl1[:,c] + h[node]·Wr1[:,c]
// N=3 -> plain VALU dot products (WMMA would waste 13/16 of the tile).
// ---------------------------------------------------------------------------
__global__ __launch_bounds__(256) void gemm1_out(
    const float* __restrict__ mean1, const float* __restrict__ h,
    const float* __restrict__ Wl1, const float* __restrict__ bl1,
    const float* __restrict__ Wr1, float* __restrict__ out) {
    int idx = blockIdx.x * blockDim.x + threadIdx.x;
    if (idx >= N_NODES * OUT_CH) return;
    int node = idx / OUT_CH;
    int c    = idx % OUT_CH;
    const float* mr = mean1 + (size_t)node * HID;
    const float* hr = h     + (size_t)node * HID;
    float acc = bl1[c];
#pragma unroll 4
    for (int k = 0; k < HID; ++k)
        acc += mr[k] * Wl1[k * OUT_CH + c] + hr[k] * Wr1[k * OUT_CH + c];
    out[idx] = acc;
}

// ---------------------------------------------------------------------------
// Launch
// ---------------------------------------------------------------------------
extern "C" void kernel_launch(void* const* d_in, const int* in_sizes, int n_in,
                              void* d_out, int out_size, void* d_ws, size_t ws_size,
                              hipStream_t stream) {
    const float* x    = (const float*)d_in[0];
    const int*   eidx = (const int*)d_in[1];
    const float* Wl0  = (const float*)d_in[2];
    const float* bl0  = (const float*)d_in[3];
    const float* Wr0  = (const float*)d_in[4];
    const float* Wl1  = (const float*)d_in[5];
    const float* bl1  = (const float*)d_in[6];
    const float* Wr1  = (const float*)d_in[7];
    float* out = (float*)d_out;

    // Workspace layout (bytes); msg0|msg1|deg contiguous for one zero pass.
    char* base = (char*)d_ws;
    float* msg0 = (float*)base;                                         // 50000*128 f32
    float* msg1 = (float*)(base + (size_t)N_NODES * IN_CH * 4);         // 50000*256 f32
    float* deg  = (float*)(base + (size_t)N_NODES * (IN_CH + HID) * 4); // 50000 f32
    float* h    = (float*)(base + (size_t)N_NODES * (IN_CH + HID + 1) * 4 + 48);
    char* p = (char*)h + (size_t)N_NODES * HID * 4;
    unsigned short* WlP = (unsigned short*)p;  p += (size_t)IN_CH * HID * 2;   // 64KB
    unsigned short* WrP = (unsigned short*)p;  p += (size_t)IN_CH * HID * 2;   // 64KB
    unsigned short* XP  = (unsigned short*)p;  p += (size_t)N_NODES * IN_CH * 2; // 12.8MB
    unsigned short* MP  = (unsigned short*)p;                                    // 12.8MB

    const size_t zeroCount = (size_t)N_NODES * (IN_CH + HID + 1);       // msg0+msg1+deg
    zero_f32<<<2048, 256, 0, stream>>>(msg0, zeroCount);

    pack_weight_bf16<<<8, 256, 0, stream>>>(Wl0, WlP);
    pack_weight_bf16<<<8, 256, 0, stream>>>(Wr0, WrP);

    // Pack x into A-fragment bf16 (independent of aggregation).
    const int nPackA = (N_NODES / 16) * 4 * 32;                          // 400000
    pack_a_bf16<<<(nPackA + 255) / 256, 256, 0, stream>>>(x, XP);

    // Layer 0 aggregate: 32 lanes/edge.
    scatter_add<IN_CH, true><<<(N_EDGES * (IN_CH / 4) + 255) / 256, 256, 0, stream>>>(
        x, eidx, msg0, deg, N_EDGES);
    mean_div<7><<<((N_NODES * IN_CH) + 255) / 256, 256, 0, stream>>>(
        msg0, deg, (size_t)N_NODES * IN_CH);
    pack_a_bf16<<<(nPackA + 255) / 256, 256, 0, stream>>>(msg0, MP);

    // Layer 0 GEMM (WMMA): 50000 tiles / 8 waves = 6250 blocks.
    gemm0_wmma<<<(N_NODES / 16) * 16 / 8, 256, 0, stream>>>(MP, XP, WlP, WrP, bl0, h);

    // Layer 1 aggregate: 64 lanes/edge.
    scatter_add<HID, false><<<(N_EDGES * (HID / 4) + 255) / 256, 256, 0, stream>>>(
        h, eidx, msg1, nullptr, N_EDGES);
    mean_div<8><<<((N_NODES * HID) + 255) / 256, 256, 0, stream>>>(
        msg1, deg, (size_t)N_NODES * HID);

    // Layer 1 GEMM (N=3, VALU).
    gemm1_out<<<(N_NODES * OUT_CH + 255) / 256, 256, 0, stream>>>(
        msg1, h, Wl1, bl1, Wr1, out);
}
